// SwinUNet_43267500540185
// MI455X (gfx1250) — matrix operand order repaired and checked by reading, not compiled, not run on hardware
//
#include <hip/hip_runtime.h>
#include <hip/hip_fp16.h>

typedef __attribute__((ext_vector_type(16))) _Float16 v16h;
typedef __attribute__((ext_vector_type(8)))  _Float16 v8h;
typedef __attribute__((ext_vector_type(4)))  _Float16 v4h;
typedef __attribute__((ext_vector_type(8)))  float    v8f;

#define WIN_N    49
#define MPAD     64
#define CDIM     384
#define C3       1152
#define NHEADS   12
#define HD       32

// LDS strides (in halfs) — padded to break bank-conflict periodicity
#define XSTR     400
#define VSTR     72
#define PSTR     72

// LDS layout (half offsets)
#define XS_OFF   0                       // 64 x 400 : X, later reused as O
#define QS_OFF   25600                   // 64 x 400 : Q (pre-scaled)
#define KS_OFF   51200                   // 64 x 400 : K
#define VT_OFF   76800                   // 384 x 72 : V transposed (dim-major)
#define PB_OFF   104448                  // 8 waves x 16 x 72 : P staging
#define LDS_HALFS 113664
#define LDS_BYTES (LDS_HALFS * 2)

// workspace byte offsets
#define WS_QKVW  0                       // 1152*384 f16
#define WS_PROJW 884736                  // 384*384 f16
#define WS_BIAS  1179648                 // 12*64*64 f32 (padded, pre-gathered)

__device__ __forceinline__ v8f wmma_f16(v16h a, v16h b, v8f c) {
  return __builtin_amdgcn_wmma_f32_16x16x32_f16(
      /*neg_a=*/false, a, /*neg_b=*/false, b,
      /*c_mod=*/(short)0, c, /*reuse_a=*/false, /*reuse_b=*/false);
}

// A-operand (16x32 f16, M x K): lane holds row m0+(lane&15); halves [0..15] at
// K = k0 + (lane>=16 ? 16 : 0) + s (16 contiguous K values).
__device__ __forceinline__ v16h load_a16(const _Float16* base, int stride,
                                         int m0, int k0, int lane) {
  int m  = m0 + (lane & 15);
  int kb = k0 + ((lane >> 4) << 4);
  const _Float16* p = base + m * stride + kb;
  v8h lo = *(const v8h*)(p);
  v8h hi = *(const v8h*)(p + 8);
  return __builtin_shufflevector(lo, hi, 0,1,2,3,4,5,6,7,8,9,10,11,12,13,14,15);
}

// B-operand (32x16 f16, K x N): lane n holds column n = 16 contiguous K values
// (caller supplies pointer already offset by n's row of B^T and khi).
__device__ __forceinline__ v16h load_b16(const _Float16* p) {
  v8h lo = *(const v8h*)(p);
  v8h hi = *(const v8h*)(p + 8);
  return __builtin_shufflevector(lo, hi, 0,1,2,3,4,5,6,7,8,9,10,11,12,13,14,15);
}

// ---------- pre-pass kernels ----------
__global__ __launch_bounds__(256) void cvt_f32_to_f16_k(const float* __restrict__ in,
                                                        _Float16* __restrict__ out, int n) {
  int i = blockIdx.x * 256 + threadIdx.x;
  if (i < n) out[i] = (_Float16)in[i];
}

__global__ __launch_bounds__(256) void build_bias64_k(const float* __restrict__ table,
                                                      const int* __restrict__ relidx,
                                                      float* __restrict__ outb) {
  int i = blockIdx.x * 256 + threadIdx.x;  // 12*64*64
  if (i >= NHEADS * 64 * 64) return;
  int h = i >> 12;
  int r = (i >> 6) & 63;
  int c = i & 63;
  float v = 0.0f;
  if (r < WIN_N && c < WIN_N) v = table[relidx[r * WIN_N + c] * NHEADS + h];
  outb[i] = v;
}

// ---------- fused window-attention kernel ----------
__global__ __launch_bounds__(256)
void swin_win_attn_k(const float* __restrict__ x,
                     const float* __restrict__ qkv_b,
                     const float* __restrict__ proj_b,
                     const _Float16* __restrict__ qkv_wh,
                     const _Float16* __restrict__ proj_wh,
                     const float* __restrict__ bias64,
                     float* __restrict__ out) {
  extern __shared__ _Float16 smem[];
  _Float16* Xs = smem + XS_OFF;   // X, later O
  _Float16* Qs = smem + QS_OFF;
  _Float16* Ks = smem + KS_OFF;
  _Float16* Vt = smem + VT_OFF;

  const int tid   = threadIdx.x;
  const int wave  = tid >> 5;
  const int lane  = tid & 31;
  const int bn    = lane & 15;          // B col / C col within tile
  const int khi   = (lane >> 4) << 4;   // +16 K offset for B loads (hi half)
  const int rbase = (lane >> 4) << 3;   // +8 row offset for C/D (hi half)

  const int win = blockIdx.x;
  const float* xw = x + (size_t)win * WIN_N * CDIM;

  // ---- Phase 0: x -> LDS f16, zero-pad rows [49,64) ----
  for (int i = tid; i < MPAD * (CDIM / 4); i += 256) {
    int r  = i / (CDIM / 4);
    int c4 = i % (CDIM / 4);
    float4 v = make_float4(0.f, 0.f, 0.f, 0.f);
    if (r < WIN_N) v = *(const float4*)(xw + r * CDIM + c4 * 4);
    v4h h = {(_Float16)v.x, (_Float16)v.y, (_Float16)v.z, (_Float16)v.w};
    *(v4h*)(Xs + r * XSTR + c4 * 4) = h;
  }
  __syncthreads();

  // ---- Phase 1: QKV = X @ W^T + b ; Q pre-scaled; V stored transposed ----
  {
    const float scale = 0.17677669529663687f;  // 1/sqrt(32)
    for (int ntl = 0; ntl < 9; ++ntl) {
      int nt  = wave * 9 + ntl;                 // 72 N-tiles over 8 waves
      int col = nt * 16 + bn;
      const _Float16* wrow = qkv_wh + (size_t)col * CDIM;
      // prefetch next N-tile's weight row while this tile computes
      if (ntl < 8) __builtin_prefetch(wrow + 16 * CDIM, 0, 1);
      float bcol = qkv_b[col];
      // wave-uniform Q/K/V selector (16-wide tiles never straddle boundaries)
      int ntu = __builtin_amdgcn_readfirstlane(nt);
      for (int mt = 0; mt < 4; ++mt) {
        v8f acc = {};
#pragma unroll
        for (int k = 0; k < 12; ++k) {
          v16h a = load_a16(Xs, XSTR, mt * 16, k * 32, lane);
          v16h b = load_b16(wrow + k * 32 + khi);
          acc = wmma_f16(a, b, acc);
        }
        union { v8f v; float f[8]; } u; u.v = acc;
        int mrow = mt * 16 + rbase;
        if (ntu < 24) {            // Q (pre-scaled)
#pragma unroll
          for (int e = 0; e < 8; ++e)
            Qs[(mrow + e) * XSTR + col] = (_Float16)((u.f[e] + bcol) * scale);
        } else if (ntu < 48) {     // K
#pragma unroll
          for (int e = 0; e < 8; ++e)
            Ks[(mrow + e) * XSTR + (col - CDIM)] = (_Float16)(u.f[e] + bcol);
        } else {                   // V, transposed (dim-major)
#pragma unroll
          for (int e = 0; e < 8; ++e)
            Vt[(col - 2 * CDIM) * VSTR + (mrow + e)] = (_Float16)(u.f[e] + bcol);
        }
      }
    }
  }
  __syncthreads();

  // ---- Phase 2: per-head attention; O written into Xs (X is dead) ----
  {
    _Float16* Pw = smem + PB_OFF + wave * (16 * PSTR);
    for (int h = wave; h < NHEADS; h += 8) {
      const int hoff = h * HD;
      for (int mt = 0; mt < 4; ++mt) {
        // S = (Q*scale) @ K^T for one 16-row block, 4 column tiles
        union { v8f v; float f[8]; } s[4];
        v16h aq = load_a16(Qs, XSTR, mt * 16, hoff, lane);
#pragma unroll
        for (int ct = 0; ct < 4; ++ct) {
          v16h b = load_b16(Ks + (ct * 16 + bn) * XSTR + hoff + khi);
          v8f z = {};
          s[ct].v = wmma_f16(aq, b, z);
        }
        // bias add + column mask + row max
        float mx[8], sm[8];
#pragma unroll
        for (int e = 0; e < 8; ++e) mx[e] = -1e30f;
#pragma unroll
        for (int ct = 0; ct < 4; ++ct) {
          int col = ct * 16 + bn;
#pragma unroll
          for (int e = 0; e < 8; ++e) {
            int m = mt * 16 + rbase + e;
            float val = s[ct].f[e] + bias64[((size_t)h * 64 + m) * 64 + col];
            if (col >= WIN_N) val = -1e30f;
            s[ct].f[e] = val;
            mx[e] = fmaxf(mx[e], val);
          }
        }
#pragma unroll
        for (int e = 0; e < 8; ++e) {
          float m = mx[e];
          for (int off = 8; off >= 1; off >>= 1)
            m = fmaxf(m, __shfl_xor(m, off));   // stays within 16-lane half
          mx[e] = m;
        }
#pragma unroll
        for (int e = 0; e < 8; ++e) sm[e] = 0.f;
#pragma unroll
        for (int ct = 0; ct < 4; ++ct) {
#pragma unroll
          for (int e = 0; e < 8; ++e) {
            float p = __expf(s[ct].f[e] - mx[e]);
            s[ct].f[e] = p;
            sm[e] += p;
          }
        }
#pragma unroll
        for (int e = 0; e < 8; ++e) {
          float t = sm[e];
          for (int off = 8; off >= 1; off >>= 1)
            t += __shfl_xor(t, off);
          sm[e] = 1.0f / t;
        }
        // stage P (16x64) in per-wave LDS tile: C-layout -> A-layout via LDS
#pragma unroll
        for (int ct = 0; ct < 4; ++ct) {
          int col = ct * 16 + bn;
#pragma unroll
          for (int e = 0; e < 8; ++e) {
            Pw[(rbase + e) * PSTR + col] = (_Float16)(s[ct].f[e] * sm[e]);
          }
        }
        __builtin_amdgcn_wave_barrier();  // LDS per-wave in-order: RAW is safe
        // O(mt, head) = P @ V  (two 16-col tiles of head dim)
#pragma unroll
        for (int dt = 0; dt < 2; ++dt) {
          v8f acc = {};
#pragma unroll
          for (int ks = 0; ks < 2; ++ks) {
            v16h a = load_a16(Pw, PSTR, 0, ks * 32, lane);
            v16h b = load_b16(Vt + (hoff + dt * 16 + bn) * VSTR + ks * 32 + khi);
            acc = wmma_f16(a, b, acc);
          }
          union { v8f v; float f[8]; } u; u.v = acc;
#pragma unroll
          for (int e = 0; e < 8; ++e) {
            int m = mt * 16 + rbase + e;
            Xs[m * XSTR + hoff + dt * 16 + bn] = (_Float16)u.f[e];
          }
        }
        __builtin_amdgcn_wave_barrier();  // Pw reused next mt
      }
    }
  }
  __syncthreads();

  // ---- Phase 3: out = O @ proj_w^T + proj_b (fp32 store, rows < 49) ----
  {
    float* outw = out + (size_t)win * WIN_N * CDIM;
    for (int ntl = 0; ntl < 3; ++ntl) {
      int nt  = wave * 3 + ntl;                 // 24 N-tiles over 8 waves
      int col = nt * 16 + bn;
      const _Float16* wrow = proj_wh + (size_t)col * CDIM;
      if (ntl < 2) __builtin_prefetch(wrow + 16 * CDIM, 0, 1);
      float bcol = proj_b[col];
#pragma unroll
      for (int mt = 0; mt < 4; ++mt) {
        v8f acc = {};
#pragma unroll
        for (int k = 0; k < 12; ++k) {
          v16h a = load_a16(Xs, XSTR, mt * 16, k * 32, lane);
          v16h b = load_b16(wrow + k * 32 + khi);
          acc = wmma_f16(a, b, acc);
        }
        union { v8f v; float f[8]; } u; u.v = acc;
        if (mt < 3) {
          // rows mt*16 .. mt*16+15 all < 49: unconditional stores
#pragma unroll
          for (int e = 0; e < 8; ++e) {
            int m = mt * 16 + rbase + e;
            outw[m * CDIM + col] = u.f[e] + bcol;
          }
        } else {
          // only row 48 is valid (VGPR 0, lanes 0-15)
          if (lane < 16) outw[48 * CDIM + col] = u.f[0] + bcol;
        }
      }
    }
  }
}

extern "C" void kernel_launch(void* const* d_in, const int* in_sizes, int n_in,
                              void* d_out, int out_size, void* d_ws, size_t ws_size,
                              hipStream_t stream) {
  const float* x          = (const float*)d_in[0];
  const float* qkv_w      = (const float*)d_in[1];
  const float* qkv_b      = (const float*)d_in[2];
  const float* proj_w     = (const float*)d_in[3];
  const float* proj_b     = (const float*)d_in[4];
  const float* bias_table = (const float*)d_in[5];
  const int*   rel_index  = (const int*)d_in[6];

  char* ws = (char*)d_ws;
  _Float16* qkv_wh  = (_Float16*)(ws + WS_QKVW);
  _Float16* proj_wh = (_Float16*)(ws + WS_PROJW);
  float*    bias64  = (float*)(ws + WS_BIAS);

  cvt_f32_to_f16_k<<<(C3 * CDIM + 255) / 256, 256, 0, stream>>>(qkv_w, qkv_wh, C3 * CDIM);
  cvt_f32_to_f16_k<<<(CDIM * CDIM + 255) / 256, 256, 0, stream>>>(proj_w, proj_wh, CDIM * CDIM);
  build_bias64_k<<<(NHEADS * 64 * 64 + 255) / 256, 256, 0, stream>>>(bias_table, rel_index, bias64);

  swin_win_attn_k<<<4096, 256, LDS_BYTES, stream>>>(x, qkv_b, proj_b, qkv_wh,
                                                    proj_wh, bias64, (float*)d_out);
}
